// LSTM_17961553232425
// MI455X (gfx1250) — matrix operand order repaired and checked by reading, not compiled, not run on hardware
//
#include <hip/hip_runtime.h>

// -------- CDNA5 WMMA / vector types --------
typedef __attribute__((ext_vector_type(16))) __bf16       v16bf;
typedef __attribute__((ext_vector_type(8)))  float        v8f;
typedef __attribute__((ext_vector_type(4)))  float        f4v;
typedef __attribute__((ext_vector_type(4)))  unsigned int u4v;

namespace {
constexpr int Bn  = 128;   // batch
constexpr int Tn  = 1024;  // time steps
constexpr int INn = 256;   // input features
constexpr int Hn  = 256;   // hidden
constexpr int H4n = 1024;  // 4*H gate width
constexpr int ROWP = 264;  // padded LDS row stride in bf16 elems (256+8) -> bank-spread

// sched_group_barrier masks
constexpr int SG_WMMA      = 0x008;
constexpr int SG_VMEM_READ = 0x020;
constexpr int SG_DS_READ   = 0x100;
}

union FragB { v16bf v; u4v q[2]; };

// 16B load with the may-alias (char) TBAA tag: out-stores in the t-loop are
// assumed to possibly alias these, which blocks LICM from hoisting 512 VGPRs
// of loop-invariant weight fragments (round-1 spill bug) -- while leaving the
// scheduler free to move them within the store-free K-loop.
__device__ __forceinline__ u4v ld16(const void* p) {
  u4v v;
  __builtin_memcpy(&v, __builtin_assume_aligned(p, 16), 16);
  return v;
}

// Fast activations: v_exp_f32 + v_rcp_f32, no IEEE division expansion.
__device__ __forceinline__ float fast_sigmoid(float x) {
  return __builtin_amdgcn_rcpf(1.0f + __expf(-x));
}
__device__ __forceinline__ float fast_tanh(float x) {
  return 2.0f * __builtin_amdgcn_rcpf(1.0f + __expf(-2.0f * x)) - 1.0f;
}

// One-time: transpose+convert weights f32 (K,4H) row-major -> bf16 (4H,K) col-major.
__global__ void lstm_cvt_weights(const float* __restrict__ wx,
                                 const float* __restrict__ wh,
                                 __bf16* __restrict__ wxT,
                                 __bf16* __restrict__ whT) {
  int idx = blockIdx.x * blockDim.x + threadIdx.x;  // 0 .. H4n*INn-1
  int col = idx >> 8;   // gate column 0..1023
  int k   = idx & 255;  // K index 0..255
  wxT[idx] = (__bf16)wx[k * H4n + col];
  whT[idx] = (__bf16)wh[k * H4n + col];
}

// Persistent scan kernel: 8 workgroups (one per 16-row batch tile), 16 waves each.
// Wave w owns H columns [16w,16w+16) and all four gates for them.
// h lives double-buffered in LDS (bf16); c lives in registers (fp32).
// Weights stream from L2 every step; sched_group_barrier directives impose a
// depth-1 software pipeline (loads of chunk kc+1 overlap WMMAs of chunk kc).
__global__ __launch_bounds__(512, 1) void lstm_scan(
    const float* __restrict__ fullx, const float* __restrict__ h0,
    const float* __restrict__ c0, const float* __restrict__ bias,
    const __bf16* wxT, const __bf16* whT,   // no restrict: may alias 'out'
    float* out) {                            // no restrict: blocks LICM hoist
  __shared__ __align__(16) __bf16 hbuf[2][16 * ROWP];

  const int btile = blockIdx.x;          // batch tile 0..7
  const int wave  = threadIdx.x >> 5;    // 0..15 -> H-column tile
  const int lane  = threadIdx.x & 31;
  const int lhalf = lane >> 4;           // 0: lanes 0-15, 1: lanes 16-31
  const int l15   = lane & 15;
  const int col   = (wave << 4) + l15;   // this lane's H column (N index)

  // Stage initial h (f32 -> bf16) into LDS buffer 0.
  for (int i = threadIdx.x; i < 16 * Hn; i += 512) {
    int r = i >> 8, cc = i & 255;
    hbuf[0][r * ROWP + cc] = (__bf16)h0[(btile * 16 + r) * Hn + cc];
  }

  // Cell state in WMMA C/D layout: VGPR r -> row M = r + 8*lhalf, col = col.
  float creg[8];
#pragma unroll
  for (int r = 0; r < 8; ++r) {
    int M = r + lhalf * 8;
    creg[r] = c0[(btile * 16 + M) * Hn + col];
  }
  float bs[4];
#pragma unroll
  for (int g = 0; g < 4; ++g) bs[g] = bias[g * Hn + col];

  // x A-fragment rows: lanes 0-15 and 16-31 both map to M = lane&15 (K halves differ).
  const float* xrow = fullx + (size_t)(btile * 16 + l15) * Tn * INn;

  // Per-lane weight fragment bases (bf16 elements). Fragment (g, kc) sits at
  // base + g*65536 + kc*32 (+16 for the second 16B half) -> imm-foldable offsets.
  const __bf16* wxlane = wxT + (size_t)col * INn + lhalf * 8;
  const __bf16* whlane = whT + (size_t)col * INn + lhalf * 8;

  __syncthreads();

  for (int t = 0; t < Tn; ++t) {
    const int rd = t & 1, wr = rd ^ 1;
    const float* xp = xrow + (size_t)t * INn;

    v8f acc[4];
#pragma unroll
    for (int g = 0; g < 4; ++g)
#pragma unroll
      for (int r = 0; r < 8; ++r) acc[g][r] = bs[g];

#pragma unroll
    for (int kc = 0; kc < 8; ++kc) {
      // 16-bit A layout: lane holds K runs [koff,koff+8) and [koff+16,koff+24).
      const int koff = kc * 32 + lhalf * 8;

      // A fragment from x_t: 16 f32 (streamed once -> non-temporal) -> bf16.
      f4v a0 = __builtin_nontemporal_load((const f4v*)(xp + koff));
      f4v a1 = __builtin_nontemporal_load((const f4v*)(xp + koff + 4));
      f4v a2 = __builtin_nontemporal_load((const f4v*)(xp + koff + 16));
      f4v a3 = __builtin_nontemporal_load((const f4v*)(xp + koff + 20));
      v16bf ax;
      ax[0]=(__bf16)a0.x;  ax[1]=(__bf16)a0.y;  ax[2]=(__bf16)a0.z;  ax[3]=(__bf16)a0.w;
      ax[4]=(__bf16)a1.x;  ax[5]=(__bf16)a1.y;  ax[6]=(__bf16)a1.z;  ax[7]=(__bf16)a1.w;
      ax[8]=(__bf16)a2.x;  ax[9]=(__bf16)a2.y;  ax[10]=(__bf16)a2.z; ax[11]=(__bf16)a2.w;
      ax[12]=(__bf16)a3.x; ax[13]=(__bf16)a3.y; ax[14]=(__bf16)a3.z; ax[15]=(__bf16)a3.w;

      // A fragment from h_{t-1} in LDS (already bf16): two 16B ds reads.
      FragB ah;
      ah.q[0] = *(const u4v*)&hbuf[rd][l15 * ROWP + koff];
      ah.q[1] = *(const u4v*)&hbuf[rd][l15 * ROWP + koff + 16];

#pragma unroll
      for (int g = 0; g < 4; ++g) {
        const int goff = g * (Hn * INn) + kc * 32;  // bf16-element offset
        FragB bx;
        bx.q[0] = ld16(wxlane + goff);
        bx.q[1] = ld16(wxlane + goff + 16);
        acc[g] = __builtin_amdgcn_wmma_f32_16x16x32_bf16(
            false, ax, false, bx.v, (short)0, acc[g], false, false);

        FragB bh;
        bh.q[0] = ld16(whlane + goff);
        bh.q[1] = ld16(whlane + goff + 16);
        acc[g] = __builtin_amdgcn_wmma_f32_16x16x32_bf16(
            false, ah.v, false, bh.v, (short)0, acc[g], false, false);
      }

      // Scheduler pipeline shape: issue this chunk's 20 VMEM reads (16 weight
      // + 4 x) and 2 LDS reads, then retire the PREVIOUS chunk's 8 WMMAs.
      // Net effect: loads run one K-chunk ahead of the WMMAs consuming them.
      __builtin_amdgcn_sched_group_barrier(SG_VMEM_READ, 20, 0);
      __builtin_amdgcn_sched_group_barrier(SG_DS_READ, 2, 0);
      if (kc > 0) __builtin_amdgcn_sched_group_barrier(SG_WMMA, 8, 0);
    }
    // Last chunk's WMMAs.
    __builtin_amdgcn_sched_group_barrier(SG_WMMA, 8, 0);

    // Gate nonlinearity + state update, entirely in the C/D register layout.
#pragma unroll
    for (int r = 0; r < 8; ++r) {
      const int M = r + lhalf * 8;
      float f = fast_sigmoid(acc[0][r]);
      float i = fast_sigmoid(acc[1][r]);
      float o = fast_sigmoid(acc[2][r]);
      float g = fast_tanh(acc[3][r]);
      float cn = creg[r] * f + g * i;
      creg[r] = cn;
      float hn = fast_tanh(cn) * o;
      // fullh is written once and never re-read by this kernel -> non-temporal.
      __builtin_nontemporal_store(
          hn, out + (((size_t)(btile * 16 + M)) * Tn + t) * Hn + col);
      hbuf[wr][M * ROWP + col] = (__bf16)hn;  // h for next step
    }
    __syncthreads();  // one barrier per step (workgroup scope only)
  }

  // Final h_f, c_f (concatenated after fullh in d_out). T even -> final h in buffer 0.
  const size_t HF = (size_t)Bn * Tn * Hn;
#pragma unroll
  for (int r = 0; r < 8; ++r) {
    const int M = r + lhalf * 8;
    const int grow = btile * 16 + M;
    out[HF + (size_t)grow * Hn + col] = (float)hbuf[0][M * ROWP + col];
    out[HF + (size_t)Bn * Hn + (size_t)grow * Hn + col] = creg[r];
  }
}

extern "C" void kernel_launch(void* const* d_in, const int* in_sizes, int n_in,
                              void* d_out, int out_size, void* d_ws, size_t ws_size,
                              hipStream_t stream) {
  const float* fullx = (const float*)d_in[0];
  const float* h0    = (const float*)d_in[1];
  const float* c0    = (const float*)d_in[2];
  const float* wx    = (const float*)d_in[3];
  const float* wh    = (const float*)d_in[4];
  const float* b     = (const float*)d_in[5];
  float* out = (float*)d_out;

  // Workspace: bf16 transposed weights, 2 * 1024 * 256 * 2B = 1 MB.
  __bf16* wxT = (__bf16*)d_ws;
  __bf16* whT = wxT + (size_t)H4n * INn;

  lstm_cvt_weights<<<(H4n * INn) / 256, 256, 0, stream>>>(wx, wh, wxT, whT);
  lstm_scan<<<Bn / 16, 512, 0, stream>>>(fullx, h0, c0, b, wxT, whT, out);
}
